// SparseFlowModel_29308856828504
// MI455X (gfx1250) — matrix operand order repaired and checked by reading, not compiled, not run on hardware
//
#include <hip/hip_runtime.h>
#include <hip/hip_bf16.h>

typedef __attribute__((ext_vector_type(16))) _Float16 v16h;
typedef __attribute__((ext_vector_type(8)))  _Float16 v8h;
typedef __attribute__((ext_vector_type(8)))  float    v8f;

#define WMMA16(A,B,C) __builtin_amdgcn_wmma_f32_16x16x32_f16(false,(A),false,(B),(short)0,(C),false,false)

// Fragment index maps (inverse of ISA VGPR layouts, 16-bit 16x32 A / 32x16 B):
//  A: k(0..31) -> hf=(k>>3)&1, e=(k&7)+8*(k>>4); lane = hf*16+row
//  B: k(0..31) -> hf=(k>>3)&1, e=(k&7)+8*(k>>4); lane = hf*16+col

__device__ __forceinline__ float gelu_tanh(float x) {
  float x3 = x * x * x;
  return 0.5f * x * (1.0f + tanhf(0.7978845608f * (x + 0.044715f * x3)));
}
__device__ __forceinline__ float silu_f(float x) {
  return x / (1.0f + __expf(-x));
}

// ---------------- t-embedding MLP (single token) ----------------
__global__ __launch_bounds__(256) void k_temb(
    const float* __restrict__ t, const float* __restrict__ w1, const float* __restrict__ b1,
    const float* __restrict__ w2, const float* __restrict__ b2, float* __restrict__ s_out)
{
  __shared__ float tin[256];
  __shared__ float x1[256];
  int c = threadIdx.x;
  float tv = t[0] * 1000.0f;
  {
    int j = c & 127;
    float fr = __expf(-9.210340371976184f * (float)j / 128.0f);
    float a = tv * fr;
    tin[c] = (c < 128) ? cosf(a) : sinf(a);
  }
  __syncthreads();
  float acc = b1[c];
  for (int k = 0; k < 256; ++k) acc += tin[k] * w1[k * 256 + c];
  x1[c] = silu_f(acc);
  __syncthreads();
  acc = b2[c];
  for (int k = 0; k < 256; ++k) acc += x1[k] * w2[k * 256 + c];
  s_out[c] = silu_f(acc);
}

// ---------------- AdaLN modulation table: mod[8][2304] ----------------
__global__ void k_mod(const float* __restrict__ s, const float* __restrict__ w,
                      const float* __restrict__ b, float* __restrict__ mod)
{
  int j = blockIdx.x * blockDim.x + threadIdx.x;
  if (j >= 8 * 2304) return;
  int i = j / 2304, c = j - i * 2304;
  float acc = b[j];
  const float* wp = w + (size_t)i * 256 * 2304 + c;
  for (int k = 0; k < 256; ++k) acc += s[k] * wp[(size_t)k * 2304];
  mod[j] = acc;
}

// ---------------- LayerNorm (+ optional per-channel modulation) ----------------
__global__ __launch_bounds__(256) void k_ln(
    const float* __restrict__ x, const float* __restrict__ sc, const float* __restrict__ sh,
    float* __restrict__ out, int N)
{
  int lane = threadIdx.x & 31;
  int token = blockIdx.x * 8 + (threadIdx.x >> 5);
  if (token >= N) return;
  const float* xp = x + (size_t)token * 256;
  float v[8], s = 0.f, s2 = 0.f;
#pragma unroll
  for (int j = 0; j < 8; ++j) { float t = xp[lane + 32 * j]; v[j] = t; s += t; s2 += t * t; }
#pragma unroll
  for (int m = 1; m < 32; m <<= 1) { s += __shfl_xor(s, m); s2 += __shfl_xor(s2, m); }
  float mean = s * (1.f / 256.f);
  float var = s2 * (1.f / 256.f) - mean * mean;
  float rstd = rsqrtf(var + 1e-6f);
  float* op = out + (size_t)token * 256;
#pragma unroll
  for (int j = 0; j < 8; ++j) {
    int c = lane + 32 * j;
    float ln = (v[j] - mean) * rstd;
    op[c] = sc ? (ln * (1.f + sc[c]) + sh[c]) : ln;
  }
}

// ---------------- RoPE on q,k inside qkv buffer (N x 768) ----------------
__global__ void k_rope(float* qkv, const int* __restrict__ coords, int N)
{
  int t = blockIdx.x * blockDim.x + threadIdx.x;
  if (t >= N * 128) return;
  int p = t & 31; int h = (t >> 5) & 3; int n = t >> 7;
  float ang = 0.f;
  if (p < 30) {
    int a = p / 10, j = p - a * 10;
    float fr = __powf(10000.0f, -(float)j / 10.0f);
    ang = (float)coords[n * 3 + a] * fr;
  }
  float cs = cosf(ang), sn = sinf(ang);
  size_t base = (size_t)n * 768 + h * 64 + 2 * p;
  float xe = qkv[base], xo = qkv[base + 1];
  qkv[base] = xe * cs - xo * sn; qkv[base + 1] = xe * sn + xo * cs;
  base += 256;
  xe = qkv[base]; xo = qkv[base + 1];
  qkv[base] = xe * cs - xo * sn; qkv[base + 1] = xe * sn + xo * cs;
}

// ---------------- 64x128-tile WMMA GEMM, ping-pong LDS, pipelined loads -------------
// EPI: 0 = acc(+bias) ; 1 = resid + gate*(acc+bias) ; 2 = gelu(acc+bias)
template<int EPI>
__global__ __launch_bounds__(256) void k_gemm(
    const float* __restrict__ A, const float* __restrict__ B,
    const float* __restrict__ bias, const float* __restrict__ gate,
    const float* resid, float* out, int M, int N, int K)
{
  __shared__ _Float16 AsF[2][4][32][24];   // [buf][mtile][lane][frag elems]
  __shared__ _Float16 BsF[2][8][32][24];   // [buf][ntile][lane][...]
  const int tid = threadIdx.x;
  const int lane = tid & 31, wv = tid >> 5;
  const int m0 = blockIdx.y * 64, n0 = blockIdx.x * 128;
  const int nt0 = (wv & 3) * 2, mta = (wv >> 2) * 2;

  // loader coords
  const int arow = tid >> 2, ak0 = (tid & 3) * 8;             // A: row, 8 consecutive k
  const int ahf = (ak0 >> 3) & 1, ae0 = (ak0 & 7) + 8 * (ak0 >> 4);
  const int alane = ahf * 16 + (arow & 15), amt = arow >> 4;
  const int bcol = tid & 127;                                  // B: col, 8 consecutive k
  const int bnt = bcol >> 4, br = bcol & 15;
  const int bk0base = (tid >> 7) * 8;

  float aReg[8];
  float bReg[2][8];

  auto gload = [&](int kk) {
    const float* ap = A + (size_t)(m0 + arow) * K + kk + ak0;
    if (kk + 32 < K) __builtin_prefetch(ap + 32, 0, 1);
#pragma unroll
    for (int j = 0; j < 8; ++j) aReg[j] = ap[j];
#pragma unroll
    for (int half = 0; half < 2; ++half) {
      int k0 = bk0base + half * 16;
      const float* bp = B + (size_t)(kk + k0) * N + n0 + bcol;
#pragma unroll
      for (int j = 0; j < 8; ++j) bReg[half][j] = bp[(size_t)j * N];
    }
  };
  auto sstore = [&](int buf) {
    v8h hv;
#pragma unroll
    for (int j = 0; j < 8; ++j) hv[j] = (_Float16)aReg[j];
    *(v8h*)&AsF[buf][amt][alane][ae0] = hv;
#pragma unroll
    for (int half = 0; half < 2; ++half) {
      int k0 = bk0base + half * 16;
      int bhf = (k0 >> 3) & 1, be0 = (k0 & 7) + 8 * (k0 >> 4);
      v8h hb;
#pragma unroll
      for (int j = 0; j < 8; ++j) hb[j] = (_Float16)bReg[half][j];
      *(v8h*)&BsF[buf][bnt][bhf * 16 + br][be0] = hb;
    }
  };

  v8f acc00 = {}, acc01 = {}, acc10 = {}, acc11 = {};
  const int nchunks = K >> 5;
  gload(0);
  sstore(0);
  for (int c = 0; c < nchunks; ++c) {
    __syncthreads();
    if (c + 1 < nchunks) gload((c + 1) * 32);   // in flight during WMMA below
    const int buf = c & 1;
    v16h a0 = *(const v16h*)&AsF[buf][mta][lane][0];
    v16h a1 = *(const v16h*)&AsF[buf][mta + 1][lane][0];
    v16h b0 = *(const v16h*)&BsF[buf][nt0][lane][0];
    v16h b1 = *(const v16h*)&BsF[buf][nt0 + 1][lane][0];
    acc00 = WMMA16(a0, b0, acc00);
    acc01 = WMMA16(a0, b1, acc01);
    acc10 = WMMA16(a1, b0, acc10);
    acc11 = WMMA16(a1, b1, acc11);
    if (c + 1 < nchunks) sstore((c + 1) & 1);
  }
  const int r = lane & 15, hf = lane >> 4;
  const int col0 = n0 + nt0 * 16 + r, col1 = col0 + 16;
  const float bv0 = bias ? bias[col0] : 0.0f;
  const float bv1 = bias ? bias[col1] : 0.0f;
#pragma unroll
  for (int v = 0; v < 8; ++v) {
    int row0 = m0 + mta * 16 + hf * 8 + v;
    int row1 = row0 + 16;
    float x00 = acc00[v] + bv0, x01 = acc01[v] + bv1;
    float x10 = acc10[v] + bv0, x11 = acc11[v] + bv1;
    if (EPI == 0) {
      out[(size_t)row0 * N + col0] = x00; out[(size_t)row0 * N + col1] = x01;
      out[(size_t)row1 * N + col0] = x10; out[(size_t)row1 * N + col1] = x11;
    } else if (EPI == 1) {
      float g0 = gate[col0], g1 = gate[col1];
      out[(size_t)row0 * N + col0] = resid[(size_t)row0 * N + col0] + g0 * x00;
      out[(size_t)row0 * N + col1] = resid[(size_t)row0 * N + col1] + g1 * x01;
      out[(size_t)row1 * N + col0] = resid[(size_t)row1 * N + col0] + g0 * x10;
      out[(size_t)row1 * N + col1] = resid[(size_t)row1 * N + col1] + g1 * x11;
    } else {
      out[(size_t)row0 * N + col0] = gelu_tanh(x00); out[(size_t)row0 * N + col1] = gelu_tanh(x01);
      out[(size_t)row1 * N + col0] = gelu_tanh(x10); out[(size_t)row1 * N + col1] = gelu_tanh(x11);
    }
  }
}

// ---------------- Windowed self-attention: block per (window, head, half) -----------
__global__ __launch_bounds__(512) void k_win_attn(
    const float* __restrict__ qkv, float* __restrict__ abuf,
    const int* __restrict__ idx, const unsigned char* __restrict__ mask, int N)
{
  __shared__ _Float16 KF[2][4][32][16];   // [dk][keytile][lane][e]  (B-frags for QK^T)
  __shared__ _Float16 VF[2][4][32][16];   // [kc2][dtile][lane][e]   (B-frags for PV)
  __shared__ float    sb[64];
  __shared__ _Float16 PF[16][2][32][16];  // per-wave P as A-frags [wave][kc2][lane][e]

  const int w = blockIdx.x, head = blockIdx.y;
  const int tid = threadIdx.x;
  const int lane = tid & 31, wv = tid >> 5;
  const int r = lane & 15, hf = lane >> 4;
  const int qbase = blockIdx.z * 256 + wv * 16;
  const int wb = w * 512;

  v16h qf[2];
  {
    int tok = idx[wb + qbase + r];
    const float* qp = qkv + (size_t)tok * 768 + head * 64;
#pragma unroll
    for (int dk = 0; dk < 2; ++dk)
#pragma unroll
      for (int e = 0; e < 16; ++e) {
        int d = dk * 32 + ((e < 8) ? (hf * 8 + e) : (16 + hf * 8 + (e - 8)));
        qf[dk][e] = (_Float16)(qp[d] * 0.125f);
      }
  }
  float mstate[8], lstate[8];
  v8f O[4] = {};
#pragma unroll
  for (int v = 0; v < 8; ++v) { mstate[v] = -3.0e38f; lstate[v] = 0.f; }

  const int lkey = tid >> 3, ld0 = (tid & 7) * 8;
  const int ldk = ld0 >> 5, lk32 = ld0 & 31;
  const int lkhf = (lk32 >> 3) & 1, lke0 = (lk32 & 7) + 8 * (lk32 >> 4);
  const int lklane = lkhf * 16 + (lkey & 15), lknt = lkey >> 4;
  const int vd = tid & 63, vkey0 = (tid >> 6) * 8;
  const int vkc2 = vkey0 >> 5, vhf = (vkey0 >> 3) & 1, ve0 = (vkey0 & 7) + 8 * ((vkey0 >> 4) & 1);
  const int vlane = vhf * 16 + (vd & 15), vdt = vd >> 4;

  for (int kc = 0; kc < 8; ++kc) {
    {
      int tok = idx[wb + kc * 64 + lkey];
      const float* kp = qkv + (size_t)tok * 768 + 256 + head * 64 + ld0;
      v8h hv;
#pragma unroll
      for (int j = 0; j < 8; ++j) hv[j] = (_Float16)kp[j];
      *(v8h*)&KF[ldk][lknt][lklane][lke0] = hv;
    }
    {
      v8h hv;
#pragma unroll
      for (int j = 0; j < 8; ++j) {
        int tok = idx[wb + kc * 64 + vkey0 + j];
        hv[j] = (_Float16)qkv[(size_t)tok * 768 + 512 + head * 64 + vd];
      }
      *(v8h*)&VF[vkc2][vdt][vlane][ve0] = hv;
    }
    if (tid < 64) sb[tid] = mask[wb + kc * 64 + tid] ? 0.f : -1.0e9f;
    __syncthreads();

    v8f S[4];
#pragma unroll
    for (int nt = 0; nt < 4; ++nt) {
      v8f s = {};
      s = WMMA16(qf[0], *(const v16h*)&KF[0][nt][lane][0], s);
      s = WMMA16(qf[1], *(const v16h*)&KF[1][nt][lane][0], s);
      float bias = sb[nt * 16 + r];
#pragma unroll
      for (int v = 0; v < 8; ++v) s[v] += bias;
      S[nt] = s;
    }
    float rsum[8], corr[8];
#pragma unroll
    for (int v = 0; v < 8; ++v) {
      float mx = fmaxf(fmaxf(S[0][v], S[1][v]), fmaxf(S[2][v], S[3][v]));
#pragma unroll
      for (int m = 1; m < 16; m <<= 1) mx = fmaxf(mx, __shfl_xor(mx, m));
      float mn = fmaxf(mstate[v], mx);
      corr[v] = __expf(mstate[v] - mn);
      mstate[v] = mn;
      rsum[v] = 0.f;
    }
#pragma unroll
    for (int nt = 0; nt < 4; ++nt) {
      int key = nt * 16 + r;
      int kc2 = key >> 5, k32 = key & 31;
      int php = (k32 >> 3) & 1, pe = (k32 & 7) + 8 * (k32 >> 4);
#pragma unroll
      for (int v = 0; v < 8; ++v) {
        float p = __expf(S[nt][v] - mstate[v]);
        rsum[v] += p;
        PF[wv][kc2][php * 16 + (v + 8 * hf)][pe] = (_Float16)p;
      }
    }
#pragma unroll
    for (int v = 0; v < 8; ++v) {
#pragma unroll
      for (int m = 1; m < 16; m <<= 1) rsum[v] += __shfl_xor(rsum[v], m);
      lstate[v] = lstate[v] * corr[v] + rsum[v];
    }
#pragma unroll
    for (int dt = 0; dt < 4; ++dt)
#pragma unroll
      for (int v = 0; v < 8; ++v) O[dt][v] *= corr[v];
    v16h pa0 = *(const v16h*)&PF[wv][0][lane][0];
    v16h pa1 = *(const v16h*)&PF[wv][1][lane][0];
#pragma unroll
    for (int dt = 0; dt < 4; ++dt) {
      O[dt] = WMMA16(pa0, *(const v16h*)&VF[0][dt][lane][0], O[dt]);
      O[dt] = WMMA16(pa1, *(const v16h*)&VF[1][dt][lane][0], O[dt]);
    }
    __syncthreads();
  }
#pragma unroll
  for (int v = 0; v < 8; ++v) {
    int qrow = qbase + hf * 8 + v;
    if (mask[wb + qrow]) {
      int tok = idx[wb + qrow];
      float* op = abuf + (size_t)tok * 256 + head * 64;
      float linv = 1.0f / lstate[v];
#pragma unroll
      for (int dt = 0; dt < 4; ++dt) op[dt * 16 + r] = O[dt][v] * linv;
    }
  }
}

// ---------------- Cross-attention: block per (64 queries, head) ---------------------
__global__ __launch_bounds__(128) void k_cross_attn(
    const float* __restrict__ q, const float* __restrict__ kv, float* __restrict__ abuf, int N)
{
  __shared__ _Float16 KF[2][4][32][16];
  __shared__ _Float16 VF[2][4][32][16];
  __shared__ _Float16 PF[4][2][32][16];
  const int head = blockIdx.y;
  const int tid = threadIdx.x;
  const int lane = tid & 31, wv = tid >> 5;
  const int r = lane & 15, hf = lane >> 4;
  const int qrow0 = blockIdx.x * 64 + wv * 16;

  v16h qf[2];
  {
    const float* qp = q + (size_t)(qrow0 + r) * 256 + head * 64;
#pragma unroll
    for (int dk = 0; dk < 2; ++dk)
#pragma unroll
      for (int e = 0; e < 16; ++e) {
        int d = dk * 32 + ((e < 8) ? (hf * 8 + e) : (16 + hf * 8 + (e - 8)));
        qf[dk][e] = (_Float16)(qp[d] * 0.125f);
      }
  }
  float mstate[8], lstate[8];
  v8f O[4] = {};
#pragma unroll
  for (int v = 0; v < 8; ++v) { mstate[v] = -3.0e38f; lstate[v] = 0.f; }

  for (int kc = 0; kc < 8; ++kc) {
#pragma unroll
    for (int it = 0; it < 4; ++it) {
      int t2 = tid + it * 128;
      int key = t2 >> 3, d0 = (t2 & 7) * 8;
      int dk = d0 >> 5, k32 = d0 & 31;
      int khf = (k32 >> 3) & 1, ke0 = (k32 & 7) + 8 * (k32 >> 4);
      const float* kp = kv + (size_t)(kc * 64 + key) * 512 + head * 64 + d0;
      v8h hv;
#pragma unroll
      for (int j = 0; j < 8; ++j) hv[j] = (_Float16)kp[j];
      *(v8h*)&KF[dk][key >> 4][khf * 16 + (key & 15)][ke0] = hv;
    }
#pragma unroll
    for (int it = 0; it < 4; ++it) {
      int t2 = tid + it * 128;
      int d = t2 & 63, key0 = (t2 >> 6) * 8;
      int kc2 = key0 >> 5, vhf = (key0 >> 3) & 1, ve0 = (key0 & 7) + 8 * ((key0 >> 4) & 1);
      v8h hv;
#pragma unroll
      for (int j = 0; j < 8; ++j)
        hv[j] = (_Float16)kv[(size_t)(kc * 64 + key0 + j) * 512 + 256 + head * 64 + d];
      *(v8h*)&VF[kc2][d >> 4][vhf * 16 + (d & 15)][ve0] = hv;
    }
    __syncthreads();
    v8f S[4];
#pragma unroll
    for (int nt = 0; nt < 4; ++nt) {
      v8f s = {};
      s = WMMA16(qf[0], *(const v16h*)&KF[0][nt][lane][0], s);
      s = WMMA16(qf[1], *(const v16h*)&KF[1][nt][lane][0], s);
      S[nt] = s;
    }
    float rsum[8], corr[8];
#pragma unroll
    for (int v = 0; v < 8; ++v) {
      float mx = fmaxf(fmaxf(S[0][v], S[1][v]), fmaxf(S[2][v], S[3][v]));
#pragma unroll
      for (int m = 1; m < 16; m <<= 1) mx = fmaxf(mx, __shfl_xor(mx, m));
      float mn = fmaxf(mstate[v], mx);
      corr[v] = __expf(mstate[v] - mn);
      mstate[v] = mn;
      rsum[v] = 0.f;
    }
#pragma unroll
    for (int nt = 0; nt < 4; ++nt) {
      int key = nt * 16 + r;
      int kc2 = key >> 5, k32 = key & 31;
      int php = (k32 >> 3) & 1, pe = (k32 & 7) + 8 * (k32 >> 4);
#pragma unroll
      for (int v = 0; v < 8; ++v) {
        float p = __expf(S[nt][v] - mstate[v]);
        rsum[v] += p;
        PF[wv][kc2][php * 16 + (v + 8 * hf)][pe] = (_Float16)p;
      }
    }
#pragma unroll
    for (int v = 0; v < 8; ++v) {
#pragma unroll
      for (int m = 1; m < 16; m <<= 1) rsum[v] += __shfl_xor(rsum[v], m);
      lstate[v] = lstate[v] * corr[v] + rsum[v];
    }
#pragma unroll
    for (int dt = 0; dt < 4; ++dt)
#pragma unroll
      for (int v = 0; v < 8; ++v) O[dt][v] *= corr[v];
    v16h pa0 = *(const v16h*)&PF[wv][0][lane][0];
    v16h pa1 = *(const v16h*)&PF[wv][1][lane][0];
#pragma unroll
    for (int dt = 0; dt < 4; ++dt) {
      O[dt] = WMMA16(pa0, *(const v16h*)&VF[0][dt][lane][0], O[dt]);
      O[dt] = WMMA16(pa1, *(const v16h*)&VF[1][dt][lane][0], O[dt]);
    }
    __syncthreads();
  }
#pragma unroll
  for (int v = 0; v < 8; ++v) {
    int row = qrow0 + hf * 8 + v;
    float* op = abuf + (size_t)row * 256 + head * 64;
    float linv = 1.0f / lstate[v];
#pragma unroll
    for (int dt = 0; dt < 4; ++dt) op[dt * 16 + r] = O[dt][v] * linv;
  }
}

extern "C" void kernel_launch(void* const* d_in, const int* in_sizes, int n_in,
                              void* d_out, int out_size, void* d_ws, size_t ws_size,
                              hipStream_t stream)
{
  const float* x_feats  = (const float*)d_in[0];
  const float* t_in     = (const float*)d_in[1];
  const float* cond     = (const float*)d_in[2];
  const float* embed_w1 = (const float*)d_in[3];
  const float* embed_w2 = (const float*)d_in[4];
  const float* embed_b2 = (const float*)d_in[5];
  const float* temb_w1  = (const float*)d_in[6];
  const float* temb_b1  = (const float*)d_in[7];
  const float* temb_w2  = (const float*)d_in[8];
  const float* temb_b2  = (const float*)d_in[9];
  const float* adaln_w  = (const float*)d_in[10];
  const float* adaln_b  = (const float*)d_in[11];
  const float* qkv_w    = (const float*)d_in[12];
  const float* qkv_b    = (const float*)d_in[13];
  const float* ao_w     = (const float*)d_in[14];
  const float* ao_b     = (const float*)d_in[15];
  const float* cq_w     = (const float*)d_in[16];
  const float* cq_b     = (const float*)d_in[17];
  const float* ckv_w    = (const float*)d_in[18];
  const float* ckv_b    = (const float*)d_in[19];
  const float* co_w     = (const float*)d_in[20];
  const float* co_b     = (const float*)d_in[21];
  const float* mlp_w1   = (const float*)d_in[22];
  const float* mlp_b1   = (const float*)d_in[23];
  const float* mlp_w2   = (const float*)d_in[24];
  const float* mlp_b2   = (const float*)d_in[25];
  const float* pred_w1  = (const float*)d_in[26];
  const float* pred_b1  = (const float*)d_in[27];
  const float* pred_w2  = (const float*)d_in[28];
  const float* pred_b2  = (const float*)d_in[29];
  const int*   coords   = (const int*)d_in[30];
  const int*   widx0    = (const int*)d_in[31];
  const unsigned char* wmask0 = (const unsigned char*)d_in[32];
  const int*   widx1    = (const int*)d_in[33];
  const unsigned char* wmask1 = (const unsigned char*)d_in[34];

  const int N   = in_sizes[0] / 512;
  const int Wn0 = in_sizes[31] / 512;
  const int Wn1 = in_sizes[33] / 512;

  char* wsb = (char*)d_ws;
  size_t off = 0;
  auto carve = [&](size_t elems) -> float* {
    float* p = (float*)(wsb + off);
    off += ((elems * 4 + 255) & ~(size_t)255);
    return p;
  };
  float* s_silu = carve(256);
  float* modb   = carve(8 * 2304);
  float* bufA   = carve((size_t)N * 1024); // qkv / cross-q / mlp hidden
  float* h      = carve((size_t)N * 256);
  float* mbuf   = carve((size_t)N * 256);
  float* abuf   = carve((size_t)N * 256);
  float* kvbuf  = carve(512 * 512);
  float* t128   = carve((size_t)N * 128);

  dim3 b256(256);
  const int lnBlocks = (N + 7) / 8;

  k_temb<<<1, 256, 0, stream>>>(t_in, temb_w1, temb_b1, temb_w2, temb_b2, s_silu);
  k_mod<<<(8 * 2304 + 255) / 256, 256, 0, stream>>>(s_silu, adaln_w, adaln_b, modb);

  // embed: h = x @ w1 @ w2 + b2
  k_gemm<0><<<dim3(1, N / 64), b256, 0, stream>>>(x_feats, embed_w1, nullptr, nullptr, nullptr, t128, N, 128, 512);
  k_gemm<0><<<dim3(2, N / 64), b256, 0, stream>>>(t128, embed_w2, embed_b2, nullptr, nullptr, h, N, 256, 128);

  for (int i = 0; i < 8; ++i) {
    const float* mod = modb + i * 2304;
    // ---- windowed self-attention ----
    k_ln<<<lnBlocks, 256, 0, stream>>>(h, mod + 256, mod + 0, mbuf, N);
    k_gemm<0><<<dim3(6, N / 64), b256, 0, stream>>>(mbuf, qkv_w + (size_t)i * 256 * 768, qkv_b + i * 768, nullptr, nullptr, bufA, N, 768, 256);
    k_rope<<<(N * 128 + 255) / 256, 256, 0, stream>>>(bufA, coords, N);
    if ((i & 1) == 0)
      k_win_attn<<<dim3(Wn0, 4, 2), 512, 0, stream>>>(bufA, abuf, widx0, wmask0, N);
    else
      k_win_attn<<<dim3(Wn1, 4, 2), 512, 0, stream>>>(bufA, abuf, widx1, wmask1, N);
    k_gemm<1><<<dim3(2, N / 64), b256, 0, stream>>>(abuf, ao_w + (size_t)i * 256 * 256, ao_b + i * 256, mod + 512, h, h, N, 256, 256);
    // ---- cross-attention ----
    k_ln<<<lnBlocks, 256, 0, stream>>>(h, mod + 1024, mod + 768, mbuf, N);
    k_gemm<0><<<dim3(2, N / 64), b256, 0, stream>>>(mbuf, cq_w + (size_t)i * 256 * 256, cq_b + i * 256, nullptr, nullptr, bufA, N, 256, 256);
    k_gemm<0><<<dim3(4, 8), b256, 0, stream>>>(cond, ckv_w + (size_t)i * 1024 * 512, ckv_b + i * 512, nullptr, nullptr, kvbuf, 512, 512, 1024);
    k_cross_attn<<<dim3(N / 64, 4), 128, 0, stream>>>(bufA, kvbuf, abuf, N);
    k_gemm<1><<<dim3(2, N / 64), b256, 0, stream>>>(abuf, co_w + (size_t)i * 256 * 256, co_b + i * 256, mod + 1280, h, h, N, 256, 256);
    // ---- MLP ----
    k_ln<<<lnBlocks, 256, 0, stream>>>(h, mod + 1792, mod + 1536, mbuf, N);
    k_gemm<2><<<dim3(8, N / 64), b256, 0, stream>>>(mbuf, mlp_w1 + (size_t)i * 256 * 1024, mlp_b1 + i * 1024, nullptr, nullptr, bufA, N, 1024, 256);
    k_gemm<1><<<dim3(2, N / 64), b256, 0, stream>>>(bufA, mlp_w2 + (size_t)i * 1024 * 256, mlp_b2 + i * 256, mod + 2048, h, h, N, 256, 1024);
  }

  // prediction head
  k_ln<<<lnBlocks, 256, 0, stream>>>(h, nullptr, nullptr, mbuf, N);
  k_gemm<0><<<dim3(1, N / 64), b256, 0, stream>>>(mbuf, pred_w1, pred_b1, nullptr, nullptr, t128, N, 128, 256);
  k_gemm<0><<<dim3(4, N / 64), b256, 0, stream>>>(t128, pred_w2, pred_b2, nullptr, nullptr, (float*)d_out, N, 512, 128);

  (void)n_in; (void)out_size; (void)ws_size;
}
